// Network_70798240907716
// MI455X (gfx1250) — matrix operand order repaired and checked by reading, not compiled, not run on hardware
//
#include <hip/hip_runtime.h>
#include <hip/hip_bf16.h>

// ---------------------------------------------------------------------------
// Pairwise MLP kernel for MI455X (gfx1250, wave32, WMMA).
// Dominant cost: 137 GFLOP fused GEMM  (B*N*M, 512) x (512, 256) with the A
// matrix generated on the fly as relu(a_proj[n] + b_proj[m] + b1).
// Path: f16 WMMA (v_wmma_f32_16x16x32_f16), f32 accumulate.
//
// W2 is pre-swizzled into WMMA B-fragment-major order so the inner j-loop
// uses only ds_load_b128 with immediate offsets (no VALU address math ->
// no WMMA->VALU WAR hazard v_nops).
// ---------------------------------------------------------------------------

#define BB   2
#define NN   1024
#define MM   256
#define DD   576
#define H1   512
#define H2   256

typedef _Float16 v16h __attribute__((ext_vector_type(16)));
typedef _Float16 v8h  __attribute__((ext_vector_type(8)));
typedef float    v8f  __attribute__((ext_vector_type(8)));

static __device__ inline v8h relu_add8(v8h a, v8h b) {
    v8h r = a + b;                         // v_pk_add_f16
    const v8h z = {};
    return __builtin_elementwise_max(r, z);  // v_pk_max_num_f16
}

static __device__ inline v16h cat8(v8h lo, v8h hi) {
    return __builtin_shufflevector(lo, hi, 0,1,2,3,4,5,6,7,8,9,10,11,12,13,14,15);
}

// ---------------------------------------------------------------------------
// Stage 1: out16[row][h] = (f16)( bias[h] + sum_d Phi[row][d] * W[d][h] )
// Phi row is block-uniform (scalar cache); W reads are fully coalesced.
// Only ~1% of total FLOPs.
// ---------------------------------------------------------------------------
__global__ __launch_bounds__(256) void proj_kernel(
    const float* __restrict__ Phi, const float* __restrict__ W,
    const float* __restrict__ bias, _Float16* __restrict__ out16)
{
    const int row = blockIdx.y;
    const int h   = blockIdx.x * 256 + threadIdx.x;
    const float* __restrict__ phi = Phi + (size_t)row * DD;
    float acc = bias ? bias[h] : 0.0f;
#pragma unroll 4
    for (int d = 0; d < DD; ++d)
        acc += phi[d] * W[(size_t)d * H1 + h];
    out16[(size_t)row * H1 + h] = (_Float16)acc;
}

// ---------------------------------------------------------------------------
// Stage 2 prep: W2 (H1 x H2 f32, row-major) -> f16 in B-fragment-major order:
//   W2S[((s*16 + j)*2 + half)*32*8 + lane*8 + e7]
// where s = k-step (K=32), j = 16-col tile, lane = ((k>>3)&1)*16 + (n%16),
// half/e7 split the 16 K-values per lane into two 16B runs.
// Matches the ISA 16-bit B layout: lanes 0-15 K=k0..+7,k0+16..+23; lanes 16-31 +8.
// ---------------------------------------------------------------------------
__global__ __launch_bounds__(256) void packW2_kernel(
    const float* __restrict__ W2, _Float16* __restrict__ W2S)
{
    const int idx = blockIdx.x * 256 + threadIdx.x;   // H1*H2 = 131072 elements
    const int k = idx >> 8;          // 0..511 (H1 row)
    const int n = idx & (H2 - 1);    // 0..255 (H2 col), coalesced reads
    const int s    = k >> 5;
    const int r    = k & 31;
    const int j    = n >> 4;
    const int l15  = n & 15;
    const int lane = (((r >> 3) & 1) << 4) | l15;
    const int e    = (r & 7) | ((r >> 4) << 3);
    const int half = e >> 3;
    const int e7   = e & 7;
    const size_t off = ((((size_t)((s * 16 + j) * 2 + half)) * 32 + lane) << 3) + e7;
    W2S[off] = (_Float16)W2[(size_t)k * H2 + n];
}

// ---------------------------------------------------------------------------
// Main fused kernel. Grid (M/16, N/16, B); block 256 = 8 waves.
// Wave w handles m in {m0+2w, m0+2w+1} x 16 n's x all 256 H2 columns:
//   32 accumulator tiles (16x16 f32) held across K=512.
// Swizzled W2 lives in LDS (256 KB), loaded once per workgroup.
// ---------------------------------------------------------------------------
__global__ __launch_bounds__(256, 1) void pair_mlp_kernel(
    const _Float16* __restrict__ aproj,   // [B*N][512]  (b1 folded in)
    const _Float16* __restrict__ bproj,   // [B*M][512]
    const _Float16* __restrict__ W2S,     // fragment-major, 131072 halves
    const float*    __restrict__ b2,      // [256]
    const float*    __restrict__ W3,      // [256]
    const float*    __restrict__ b3,      // [1]
    float*          __restrict__ out)     // [B][N][M]
{
    extern __shared__ _Float16 lds[];     // 131072 halves = 256 KB

    const int tid = threadIdx.x;

    // ---- stage swizzled W2 into LDS: flat coalesced 16B-chunk copy ----
    for (int c = tid; c < (H1 * H2) / 8; c += 256)
        *(v8h*)&lds[(size_t)c * 8] = *(const v8h*)&W2S[(size_t)c * 8];
    __syncthreads();

    const int b     = blockIdx.z;
    const int n0    = blockIdx.y << 4;
    const int m0    = blockIdx.x << 4;
    const int wave  = tid >> 5;
    const int lane  = tid & 31;
    const int lhalf = lane >> 4;          // 0: lanes 0-15, 1: lanes 16-31
    const int l15   = lane & 15;

    const int mA = m0 + wave * 2;         // two m's per wave

    // A-side pointers (global; hot in WGP$/L2 since 16 rows shared by all waves)
    const _Float16* __restrict__ aRow  = aproj + ((size_t)(b * NN + n0 + l15)) * H1;
    const _Float16* __restrict__ bRow0 = bproj + ((size_t)(b * MM + mA)) * H1;
    const _Float16* __restrict__ bRow1 = bRow0 + H1;
    const int koffA = lhalf * 8;          // ISA A layout: K runs {koff..+7, koff+16..+23}

    v8f acc[2][16];
    const v8f vzero = {0.f,0.f,0.f,0.f,0.f,0.f,0.f,0.f};
#pragma unroll
    for (int a = 0; a < 2; ++a)
#pragma unroll
        for (int j = 0; j < 16; ++j) acc[a][j] = vzero;

    // per-lane fragment base (halves); per k-step stride = 8192 halves (16 KB)
    const _Float16* fragBase = &lds[(size_t)lane * 8];

    // -------------------- K loop: 16 steps of K=32 --------------------
#pragma unroll 1
    for (int s = 0; s < 16; ++s) {
        const int k0 = s * 32;

        const v8h aLo  = *(const v8h*)&aRow[k0 + koffA];
        const v8h aHi  = *(const v8h*)&aRow[k0 + koffA + 16];
        const v8h b0Lo = *(const v8h*)&bRow0[k0 + koffA];
        const v8h b0Hi = *(const v8h*)&bRow0[k0 + koffA + 16];
        const v8h b1Lo = *(const v8h*)&bRow1[k0 + koffA];
        const v8h b1Hi = *(const v8h*)&bRow1[k0 + koffA + 16];

        // h1 = relu(a_proj + b_proj (+b1 already folded)) built in-register
        const v16h af0 = cat8(relu_add8(aLo, b0Lo), relu_add8(aHi, b0Hi));
        const v16h af1 = cat8(relu_add8(aLo, b1Lo), relu_add8(aHi, b1Hi));

        const _Float16* fb = fragBase + (size_t)s * 8192;
#pragma unroll
        for (int j = 0; j < 16; ++j) {
            // both loads: base VGPR + immediate offset (j*1024 / +512 bytes)
            const v8h blo = *(const v8h*)(fb + j * 512);
            const v8h bhi = *(const v8h*)(fb + j * 512 + 256);
            const v16h bf = cat8(blo, bhi);
            acc[0][j] = __builtin_amdgcn_wmma_f32_16x16x32_f16(
                false, af0, false, bf, (short)0, acc[0][j], false, false);
            acc[1][j] = __builtin_amdgcn_wmma_f32_16x16x32_f16(
                false, af1, false, bf, (short)0, acc[1][j], false, false);
        }
    }

    // -------------------- epilogue: relu(h2+b2) . W3 + b3 --------------------
    float rs0[8], rs1[8];
#pragma unroll
    for (int g = 0; g < 8; ++g) { rs0[g] = 0.f; rs1[g] = 0.f; }

#pragma unroll
    for (int j = 0; j < 16; ++j) {
        const int col = j * 16 + l15;
        const float w3 = W3[col];
        const float bc = b2[col];
#pragma unroll
        for (int g = 0; g < 8; ++g) {
            float c0 = acc[0][j][g] + bc; c0 = fmaxf(c0, 0.f);
            float c1 = acc[1][j][g] + bc; c1 = fmaxf(c1, 0.f);
            rs0[g] += c0 * w3;
            rs1[g] += c1 * w3;
        }
    }

    // butterfly-sum the 16 column-lanes of each half-wave
#pragma unroll
    for (int g = 0; g < 8; ++g) {
        float v0 = rs0[g], v1 = rs1[g];
#pragma unroll
        for (int off = 1; off < 16; off <<= 1) {
            v0 += __shfl_xor(v0, off);
            v1 += __shfl_xor(v1, off);
        }
        rs0[g] = v0; rs1[g] = v1;
    }

    if (l15 == 0) {                       // lanes 0 and 16 write 8 rows each
        const float bb = b3[0];
#pragma unroll
        for (int g = 0; g < 8; ++g) {
            const int n = n0 + g + lhalf * 8;   // tile rows: VGPR g -> M=g / g+8
            out[((size_t)b * NN + n) * MM + (mA + 0)] = rs0[g] + bb;
            out[((size_t)b * NN + n) * MM + (mA + 1)] = rs1[g] + bb;
        }
    }
}

// ---------------------------------------------------------------------------
extern "C" void kernel_launch(void* const* d_in, const int* in_sizes, int n_in,
                              void* d_out, int out_size, void* d_ws, size_t ws_size,
                              hipStream_t stream)
{
    const float* PhiA = (const float*)d_in[0];
    const float* PhiB = (const float*)d_in[1];
    const float* W1a  = (const float*)d_in[2];
    const float* W1b  = (const float*)d_in[3];
    const float* b1   = (const float*)d_in[4];
    const float* W2   = (const float*)d_in[5];
    const float* b2   = (const float*)d_in[6];
    const float* W3   = (const float*)d_in[7];
    const float* b3   = (const float*)d_in[8];
    float* out = (float*)d_out;

    // workspace: aproj16 | bproj16 | W2S  (f16), ~2.75 MB total
    _Float16* aproj16 = (_Float16*)d_ws;
    _Float16* bproj16 = aproj16 + (size_t)BB * NN * H1;
    _Float16* W2S     = bproj16 + (size_t)BB * MM * H1;

    // Stage 1: projections (b1 folded into aproj)
    proj_kernel<<<dim3(H1 / 256, BB * NN), 256, 0, stream>>>(PhiA, W1a, b1, aproj16);
    proj_kernel<<<dim3(H1 / 256, BB * MM), 256, 0, stream>>>(PhiB, W1b, nullptr, bproj16);

    // Stage 2 prep: W2 -> f16, WMMA-fragment-major
    packW2_kernel<<<dim3((H1 * H2) / 256), 256, 0, stream>>>(W2, W2S);

    // Main fused WMMA kernel: 256 KB dynamic LDS (within 320 KB/WGP)
    const size_t ldsBytes = (size_t)H1 * H2 * sizeof(_Float16);  // 262144
    hipFuncSetAttribute((const void*)pair_mlp_kernel,
                        hipFuncAttributeMaxDynamicSharedMemorySize, (int)ldsBytes);
    pair_mlp_kernel<<<dim3(MM / 16, NN / 16, BB), 256, ldsBytes, stream>>>(
        aproj16, bproj16, W2S, b2, W3, b3, out);
}
